// NLDS_15753940042018
// MI455X (gfx1250) — matrix operand order repaired and checked by reading, not compiled, not run on hardware
//
#include <hip/hip_runtime.h>

typedef unsigned short u16t;
typedef __attribute__((ext_vector_type(4)))  u16t   v4us;
typedef __attribute__((ext_vector_type(8)))  u16t   v8us;
typedef __attribute__((ext_vector_type(16))) u16t   v16us;
typedef __attribute__((ext_vector_type(16))) __bf16 v16bf;
typedef __attribute__((ext_vector_type(8)))  float  v8f;
typedef __attribute__((ext_vector_type(4)))  float  v4f;
typedef __attribute__((ext_vector_type(4)))  int    v4i;

// Async-to-LDS path (CDNA5): guarded so compilation always succeeds.
#if defined(__has_builtin)
#  if __has_builtin(__builtin_amdgcn_global_load_async_to_lds_b128) && \
      __has_builtin(__builtin_amdgcn_s_wait_asynccnt)
#    define NLDS_ASYNC 1
#  endif
#endif
#ifndef NLDS_ASYNC
#  define NLDS_ASYNC 0
#endif

#if NLDS_ASYNC
// Signature (from clang diagnostic): arg0 = v4i addrspace(1)* (non-const),
// arg1 = LDS-side pointer, arg2/arg3 = imm offset / cpol.
typedef __attribute__((address_space(1))) v4i* nlds_gp;
typedef __attribute__((address_space(3))) v4i* nlds_lp;
#endif

// Problem sizes (fixed by the reference)
constexpr int BSZ = 8, SEQ = 2048, DSTATE = 1024;
constexpr int MROWS = BSZ * SEQ;      // 16384
constexpr int KDIM  = 1024;
constexpr int NDIM  = 1024;

// GEMM tiling: 256x128 block, 8 waves of 64x64, BK=32 (one WMMA K-step)
constexpr int BM = 256, BN = 128, BK = 32;
constexpr int LDT = BK + 8;           // u16 stride (40) -> 80B rows, 16B-aligned slots
constexpr int KTILES = KDIM / BK;     // 32

// Scan decomposition
constexpr int NC = 64;                // chunks along T
constexpr int CL = SEQ / NC;          // 32 elements per chunk

__device__ __forceinline__ u16t f2bf(float f) {
  unsigned int u = __builtin_bit_cast(unsigned int, f);
  u += 0x7FFFu + ((u >> 16) & 1u);    // round-to-nearest-even
  return (u16t)(u >> 16);
}

// ---------- f32 -> bf16 pre-conversion (8 elements / thread, b128 stores) ----------
__global__ __launch_bounds__(256)
void NLDS_cvt_bf16(const float* __restrict__ src, u16t* __restrict__ dst) {
  size_t i = (size_t)blockIdx.x * blockDim.x + threadIdx.x;
  const v4f* s = (const v4f*)(src + i * 8);
  v4f a = s[0], b = s[1];
  v8us o = { f2bf(a.x), f2bf(a.y), f2bf(a.z), f2bf(a.w),
             f2bf(b.x), f2bf(b.y), f2bf(b.z), f2bf(b.w) };
  *(v8us*)(dst + i * 8) = o;
}

// ---------- C[M,N](f32) = A[M,K](bf16) * B[K,N](bf16), f32 accumulate ----------
__global__ __launch_bounds__(256)
void NLDS_gemm_bf16(const u16t* __restrict__ Agl,
                    const u16t* __restrict__ Bgl,
                    float* __restrict__ Cgl) {
  __shared__ __align__(16) u16t lA[2][BM * LDT];   // [m][k] row-major
  __shared__ __align__(16) u16t lB[2][BN * LDT];   // [n][k] transposed tile

  const int tid  = threadIdx.x;
  const int lane = tid & 31;
  const int wave = tid >> 5;                    // 0..7  -> 4(M) x 2(N)
  const int wm   = (wave >> 1) * 64;            // 0,64,128,192
  const int wn   = (wave & 1)  * 64;            // 0,64
  const int l15  = lane & 15;
  const int halo = lane >> 4;                   // 0 = lanes 0-15, 1 = lanes 16-31

  const int blockRow = blockIdx.y * BM;
  const int blockCol = blockIdx.x * BN;

  v8f acc[4][4];
  const v8f vzero = {0.f, 0.f, 0.f, 0.f, 0.f, 0.f, 0.f, 0.f};
#pragma unroll
  for (int mi = 0; mi < 4; ++mi)
#pragma unroll
    for (int ni = 0; ni < 4; ++ni) acc[mi][ni] = vzero;

  // Stage A tile (256x32 bf16 = 16KB): 4x b128 chunks per thread, async if available.
  auto stageA = [&](int k0, int p) {
#pragma unroll
    for (int i = 0; i < 4; ++i) {
      int idx = tid + i * 256;                  // 1024 chunks of 16B
      int row = idx >> 2;                       // 0..255
      int ch  = (idx & 3) * 8;                  // u16 offset within row
      const u16t* g = Agl + (size_t)(blockRow + row) * KDIM + k0 + ch;
      u16t* l = &lA[p][row * LDT + ch];
#if NLDS_ASYNC
      __builtin_amdgcn_global_load_async_to_lds_b128((nlds_gp)g, (nlds_lp)l, 0, 0);
#else
      *(v8us*)l = *(const v8us*)g;
#endif
    }
  };

  // Stage B tile (32x128 bf16) transposed: 4x4 register transpose, 4x ds_store_b64.
  auto stageB = [&](int k0, int p) {
    int nb = (tid & 31) * 4;                    // n block base 0..124
    int kb = (tid >> 5) * 4;                    // k block base 0..28
    v4us r[4];
#pragma unroll
    for (int j = 0; j < 4; ++j)
      r[j] = *(const v4us*)(Bgl + (size_t)(k0 + kb + j) * NDIM + blockCol + nb);
#pragma unroll
    for (int n = 0; n < 4; ++n) {
      v4us c = { r[0][n], r[1][n], r[2][n], r[3][n] };
      *(v4us*)&lB[p][(nb + n) * LDT + kb] = c;
    }
  };

  stageA(0, 0);
  stageB(0, 0);

  for (int kt = 0; kt < KTILES; ++kt) {
    int p = kt & 1;
#if NLDS_ASYNC
    __builtin_amdgcn_s_wait_asynccnt(0);
#endif
    __syncthreads();                            // buffer p fully staged, 1-p free
    if (kt + 1 < KTILES) {
      stageA((kt + 1) * BK, p ^ 1);
      stageB((kt + 1) * BK, p ^ 1);
    }

    // B fragments: lane<16 holds K=0..15 of column n; lane>=16 holds K=16..31.
    v16bf bfrag[4];
#pragma unroll
    for (int ni = 0; ni < 4; ++ni) {
      int n  = wn + ni * 16 + l15;
      int kb = halo * 16;
      v16us t;
      ((v8us*)&t)[0] = *(const v8us*)&lB[p][n * LDT + kb];
      ((v8us*)&t)[1] = *(const v8us*)&lB[p][n * LDT + kb + 8];
      bfrag[ni] = __builtin_bit_cast(v16bf, t);
    }
    // A fragments + WMMA: lane<16 = K 0..15, lane>=16 = K 16..31.
#pragma unroll
    for (int mi = 0; mi < 4; ++mi) {
      int r  = wm + mi * 16 + l15;
      int kb = halo * 16;
      v16us t;
      ((v8us*)&t)[0] = *(const v8us*)&lA[p][r * LDT + kb];
      ((v8us*)&t)[1] = *(const v8us*)&lA[p][r * LDT + kb + 8];
      v16bf afrag = __builtin_bit_cast(v16bf, t);
#pragma unroll
      for (int ni = 0; ni < 4; ++ni) {
        acc[mi][ni] = __builtin_amdgcn_wmma_f32_16x16x32_bf16(
            false, afrag, false, bfrag[ni], (short)0, acc[mi][ni], false, false);
      }
    }
  }

  // Store: v8f element r lives at M = r + 8*halo, N = l15 within each 16x16 tile.
#pragma unroll
  for (int mi = 0; mi < 4; ++mi)
#pragma unroll
    for (int ni = 0; ni < 4; ++ni) {
      int col = blockCol + wn + ni * 16 + l15;
#pragma unroll
      for (int r = 0; r < 8; ++r) {
        int row = blockRow + wm + mi * 16 + halo * 8 + r;
        Cgl[(size_t)row * NDIM + col] = acc[mi][ni][r];
      }
    }
}

// ---------- Phase 1: local scans per (b, chunk, s), zero carry-in ----------
__global__ __launch_bounds__(256)
void NLDS_scan_local(const float* __restrict__ Avec,
                     float* __restrict__ u,
                     float* __restrict__ carry) {
  int g = blockIdx.x * blockDim.x + threadIdx.x;   // [b][c][s]
  int s = g & (DSTATE - 1);
  int c = (g >> 10) & (NC - 1);
  int b = g >> 16;
  float a = Avec[s];
  float h = 0.f;
  size_t idx = ((size_t)b * SEQ + (size_t)c * CL) * DSTATE + s;
#pragma unroll 4
  for (int j = 0; j < CL; ++j) {
    h = fmaf(h, a, u[idx]);
    u[idx] = h;
    idx += DSTATE;
  }
  carry[(size_t)g] = h;
}

// ---------- Phase 2: combine chunk carries: P_{c+1} = a^CL * P_c + carry_c ----------
__global__ __launch_bounds__(256)
void NLDS_scan_carry(const float* __restrict__ Avec,
                     const float* __restrict__ h0,
                     float* __restrict__ carry) {
  int g = blockIdx.x * blockDim.x + threadIdx.x;   // b*DSTATE + s
  int s = g & (DSTATE - 1);
  int b = g >> 10;
  float a = Avec[s];
  float aL = a;
#pragma unroll
  for (int i = 0; i < 5; ++i) aL *= aL;            // a^32 == a^CL
  float P = h0[s];
  size_t base = (size_t)b * NC * DSTATE + s;
  for (int c = 0; c < NC; ++c) {
    float cy = carry[base + (size_t)c * DSTATE];
    carry[base + (size_t)c * DSTATE] = P;
    P = fmaf(aL, P, cy);
  }
}

// ---------- Phase 3: apply carry, emit bf16 hs: h = local + a^(j+1) * P_c ----------
__global__ __launch_bounds__(256)
void NLDS_scan_apply(const float* __restrict__ Avec,
                     const float* __restrict__ carry,
                     const float* __restrict__ u,
                     u16t* __restrict__ hs) {
  int g = blockIdx.x * blockDim.x + threadIdx.x;
  int s = g & (DSTATE - 1);
  int c = (g >> 10) & (NC - 1);
  int b = g >> 16;
  float a = Avec[s];
  float P = carry[(size_t)g];
  float f = a;
  size_t idx = ((size_t)b * SEQ + (size_t)c * CL) * DSTATE + s;
#pragma unroll 4
  for (int j = 0; j < CL; ++j) {
    hs[idx] = f2bf(fmaf(f, P, u[idx]));
    f *= a;
    idx += DSTATE;
  }
}

extern "C" void kernel_launch(void* const* d_in, const int* in_sizes, int n_in,
                              void* d_out, int out_size, void* d_ws, size_t ws_size,
                              hipStream_t stream) {
  const float* X   = (const float*)d_in[0];   // [8, 2048, 1024]
  const float* A   = (const float*)d_in[1];   // [1024]
  const float* Bm  = (const float*)d_in[2];   // [1024, 1024]
  const float* Cm  = (const float*)d_in[3];   // [1024, 1024]
  const float* h0  = (const float*)d_in[4];   // [1024]
  float* out = (float*)d_out;                 // [8, 2048, 1024]

  // Workspace layout (floats / u16)
  float* u      = (float*)d_ws;                               // 64 MiB f32
  float* carry  = u + (size_t)MROWS * DSTATE;                 // 2 MiB f32
  u16t*  Xbf    = (u16t*)(carry + (size_t)BSZ * NC * DSTATE); // 32 MiB bf16 (reused as hs)
  u16t*  Bmbf   = Xbf  + (size_t)MROWS * KDIM;                // 2 MiB bf16
  u16t*  Cbf    = Bmbf + (size_t)KDIM * NDIM;                 // 2 MiB bf16
  u16t*  hsbf   = Xbf;                                        // alias: X consumed by GEMM1

  // 0) pre-convert operands to bf16
  NLDS_cvt_bf16<<<(MROWS * KDIM / 8) / 256, 256, 0, stream>>>(X,  Xbf);
  NLDS_cvt_bf16<<<(KDIM * NDIM / 8) / 256, 256, 0, stream>>>(Bm, Bmbf);
  NLDS_cvt_bf16<<<(KDIM * NDIM / 8) / 256, 256, 0, stream>>>(Cm, Cbf);

  dim3 gridGemm(NDIM / BN, MROWS / BM);                       // (8, 64)

  // 1) u = X @ Bm
  NLDS_gemm_bf16<<<gridGemm, 256, 0, stream>>>(Xbf, Bmbf, u);

  // 2) chunked diagonal scan; emits bf16 hs into Xbf's space
  int nScan = BSZ * NC * DSTATE;                              // 524288
  NLDS_scan_local<<<nScan / 256, 256, 0, stream>>>(A, u, carry);
  NLDS_scan_carry<<<(BSZ * DSTATE) / 256, 256, 0, stream>>>(A, h0, carry);
  NLDS_scan_apply<<<nScan / 256, 256, 0, stream>>>(A, carry, u, hsbf);

  // 3) out = hs @ C
  NLDS_gemm_bf16<<<gridGemm, 256, 0, stream>>>(hsbf, Cbf, out);
}
